// Feature_Fusion_Net_71356586656365
// MI455X (gfx1250) — compile-verified, matrix operand order checked by reading, not run on hardware
//
#include <hip/hip_runtime.h>

// CDNA5 / gfx1250 wave32 WMMA types
typedef __attribute__((ext_vector_type(4)))  __bf16 v4bf;
typedef __attribute__((ext_vector_type(8)))  __bf16 v8bf;
typedef __attribute__((ext_vector_type(16))) __bf16 v16bf;  // A/B frag: 16x32 bf16 = 8 VGPRs/lane
typedef __attribute__((ext_vector_type(8)))  float  v8f;    // C/D frag: 16x16 f32 = 8 VGPRs/lane

#define M_DIM 1024
#define N_DIM 2048
#define K_DIM 8192
#define BM 128
#define BN 128
#define KB   64   // K staged per block, sync path (2 k-steps of 32)
#define STR  72   // sync path LDS row: 64+8 bf16 -> 144B stride, conflict-free b128 frags
#define KB2  32   // K staged per buffer, async double-buffered path
#define STR2 40   // async path LDS row: 32+8 bf16 -> 80B stride, conflict-free b128 frags

// Async-to-LDS staging only if this toolchain exposes the gfx1250 builtins
// (evaluated in the device pass; host pass harmlessly parses the sync branch).
#if __has_builtin(__builtin_amdgcn_global_load_async_to_lds_b128) && \
    __has_builtin(__builtin_amdgcn_s_wait_asynccnt)
#define ASYNC_STAGING 1
#else
#define ASYNC_STAGING 0
#endif

// Per-column BN affine terms (fully rewritten every launch -> deterministic).
__device__ float g_scale[N_DIM];
__device__ float g_shift[N_DIM];

// Load one 16x32 bf16 fragment row-slice for this lane from an LDS tile.
// v16bf layout (ISA 7.12.2, 16-bit A 16x32): lo lanes [K0..7 | K16..23],
// hi lanes [K8..15 | K24..31] -> two contiguous 16B chunks at koff and koff+16.
template <int STRIDE>
__device__ inline v16bf frag_load(const __bf16* __restrict__ tile, int row, int koff)
{
    union { v16bf v; v8bf h[2]; } u;
    const __bf16* p = tile + row * STRIDE + koff;
    u.h[0] = *(const v8bf*)(p);
    u.h[1] = *(const v8bf*)(p + 16);
    return u.v;
}

// One 32-wide k-step of the 3xBF16 scheme: 24 v_wmma_f32_16x16x32_bf16,
//   acc += ah*bh + al*bh + ah*bl   (fp32 accumulate).
template <int STRIDE>
__device__ inline void compute_step(const __bf16* __restrict__ As_hi,
                                    const __bf16* __restrict__ As_lo,
                                    const __bf16* __restrict__ Bs_hi,
                                    const __bf16* __restrict__ Bs_lo,
                                    int wm, int wn, int lr, int kb,
                                    v8f (&acc)[2][4])
{
    v16bf ah[2], al[2];
#pragma unroll
    for (int mi = 0; mi < 2; ++mi) {
        ah[mi] = frag_load<STRIDE>(As_hi, wm + mi * 16 + lr, kb);
        al[mi] = frag_load<STRIDE>(As_lo, wm + mi * 16 + lr, kb);
    }
#pragma unroll
    for (int ni = 0; ni < 4; ++ni) {
        const v16bf bh = frag_load<STRIDE>(Bs_hi, wn + ni * 16 + lr, kb);
        const v16bf bl = frag_load<STRIDE>(Bs_lo, wn + ni * 16 + lr, kb);
#pragma unroll
        for (int mi = 0; mi < 2; ++mi) {
            acc[mi][ni] = __builtin_amdgcn_wmma_f32_16x16x32_bf16(
                false, ah[mi], false, bh, (short)0, acc[mi][ni], false, false);
            acc[mi][ni] = __builtin_amdgcn_wmma_f32_16x16x32_bf16(
                false, al[mi], false, bh, (short)0, acc[mi][ni], false, false);
            acc[mi][ni] = __builtin_amdgcn_wmma_f32_16x16x32_bf16(
                false, ah[mi], false, bl, (short)0, acc[mi][ni], false, false);
        }
    }
}

#define GEMM_EPILOGUE()                                                             \
    _Pragma("unroll")                                                               \
    for (int mi = 0; mi < 2; ++mi)                                                  \
        _Pragma("unroll")                                                           \
        for (int ni = 0; ni < 4; ++ni) {                                            \
            const int col = n0 + wn + ni * 16 + lr;                                 \
            _Pragma("unroll")                                                       \
            for (int v = 0; v < 8; ++v) {                                           \
                const int row = m0 + wm + mi * 16 + hi * 8 + v;                     \
                Y[(size_t)row * N_DIM + col] = acc[mi][ni][v];                      \
            }                                                                       \
        }

#if ASYNC_STAGING
// Builtin signature (from the toolchain's diagnostic): first param is
// 'int __attribute__((vector_size(16))) __device__ *' (global AS1 int4*),
// then the LDS (AS3) destination, then imm offset and imm cpol.
typedef int v4i_vs __attribute__((vector_size(16)));
typedef __attribute__((address_space(1))) v4i_vs as1_v4i;
typedef __attribute__((address_space(3))) v4i_vs as3_v4i;

// One 16B async global->LDS transfer per lane (GLOBAL_LOAD_ASYNC_TO_LDS_B128,
// tracked by ASYNCcnt). Casts via inttoptr: AS1 == flat address numerically,
// AS3 == low-32 truncation per the ISA LDS aperture rule.
__device__ inline void async16(const __bf16* g, __bf16* l)
{
    __builtin_amdgcn_global_load_async_to_lds_b128(
        (as1_v4i*)(unsigned long long)(uintptr_t)g,
        (as3_v4i*)(unsigned long long)(uintptr_t)l,
        0, 0);
}
#endif

// ---------------------------------------------------------------------------
// Pass 0: split fp32 -> bf16 hi/lo planes (once per operand; removes all
// conversion VALU from the GEMM inner loop).
// ---------------------------------------------------------------------------
__global__ __launch_bounds__(256)
void split_bf16(const float* __restrict__ src,
                __bf16* __restrict__ dhi, __bf16* __restrict__ dlo)
{
    const size_t i = (size_t)(blockIdx.x * blockDim.x + threadIdx.x) * 4;
    float4 v = *(const float4*)(src + i);
    __bf16 h0 = (__bf16)v.x, h1 = (__bf16)v.y, h2 = (__bf16)v.z, h3 = (__bf16)v.w;
    v4bf hv = {h0, h1, h2, h3};
    v4bf lv = {(__bf16)(v.x - (float)h0), (__bf16)(v.y - (float)h1),
               (__bf16)(v.z - (float)h2), (__bf16)(v.w - (float)h3)};
    *(v4bf*)(dhi + i) = hv;
    *(v4bf*)(dlo + i) = lv;
}

// ---------------------------------------------------------------------------
// GEMM, pre-split operands. 8 waves, 128x128 tile, wave owns 32x64 (2x4 acc).
// Async toolchains: double-buffered LDS fed by async b128 transfers.
// Otherwise: single-buffered b128 copy staging.
// ---------------------------------------------------------------------------
__global__ __launch_bounds__(256)
void gemm_wmma_bf16x3_pre(const __bf16* __restrict__ Xhi, const __bf16* __restrict__ Xlo,
                          const __bf16* __restrict__ Whi, const __bf16* __restrict__ Wlo,
                          float* __restrict__ Ybase, int kChunk)
{
    const int tid  = threadIdx.x;
    const int m0   = blockIdx.y * BM;
    const int n0   = blockIdx.x * BN;
    const int kBeg = blockIdx.z * kChunk;
    const int kEnd = kBeg + kChunk;
    float* __restrict__ Y = Ybase + (size_t)blockIdx.z * (size_t)M_DIM * N_DIM;

    const int wave = tid >> 5;
    const int lane = tid & 31;
    const int lr   = lane & 15;
    const int hi   = lane >> 4;
    const int wm   = (wave & 3) * 32;
    const int wn   = (wave >> 2) * 64;

    v8f zero = {};
    v8f acc[2][4];
#pragma unroll
    for (int mi = 0; mi < 2; ++mi)
#pragma unroll
        for (int ni = 0; ni < 4; ++ni) acc[mi][ni] = zero;

#if ASYNC_STAGING
    __shared__ __bf16 As_hi[2][BM * STR2];
    __shared__ __bf16 As_lo[2][BM * STR2];
    __shared__ __bf16 Bs_hi[2][BN * STR2];
    __shared__ __bf16 Bs_lo[2][BN * STR2];

    // 4 threads per 32-bf16 row (16B = b128 per thread), 64 rows per pass.
    const int r0 = tid >> 2;        // 0..63
    const int c8 = (tid & 3) * 8;   // bf16 col 0..24

    auto stage = [&](int buf, int kk) {
#pragma unroll
        for (int j = 0; j < 2; ++j) {
            const int row = j * 64 + r0;
            const size_t ga = (size_t)(m0 + row) * K_DIM + kk + c8;
            const size_t gb = (size_t)(n0 + row) * K_DIM + kk + c8;
            const int    ls = row * STR2 + c8;
            async16(Xhi + ga, &As_hi[buf][ls]);
            async16(Xlo + ga, &As_lo[buf][ls]);
            async16(Whi + gb, &Bs_hi[buf][ls]);
            async16(Wlo + gb, &Bs_lo[buf][ls]);
        }
    };

    stage(0, kBeg);                       // 8 asyncs outstanding
    int it = 0;
    for (int kk = kBeg; kk < kEnd; kk += KB2, ++it) {
        const int  cur  = it & 1;
        const bool more = (kk + KB2) < kEnd;
        if (more) stage(cur ^ 1, kk + KB2);            // buf freed by last iter's 2nd barrier
        if (more) __builtin_amdgcn_s_wait_asynccnt(8); // in-order: cur's 8 are done
        else      __builtin_amdgcn_s_wait_asynccnt(0);
        __syncthreads();
        compute_step<STR2>(As_hi[cur], As_lo[cur], Bs_hi[cur], Bs_lo[cur],
                           wm, wn, lr, 8 * hi, acc);
        __syncthreads();
    }
#else
    __shared__ __bf16 As_hi[BM * STR];
    __shared__ __bf16 As_lo[BM * STR];
    __shared__ __bf16 Bs_hi[BN * STR];
    __shared__ __bf16 Bs_lo[BN * STR];

    // 8 threads per 64-bf16 row (16B = v8bf per thread), 32 rows per pass.
    const int r0 = tid >> 3;        // 0..31
    const int c8 = (tid & 7) * 8;   // bf16 col 0..56

    for (int kk = kBeg; kk < kEnd; kk += KB) {
#pragma unroll
        for (int j = 0; j < 4; ++j) {
            const int row = j * 32 + r0;
            const size_t ga = (size_t)(m0 + row) * K_DIM + kk + c8;
            const size_t gb = (size_t)(n0 + row) * K_DIM + kk + c8;
            const int    ls = row * STR + c8;
            *(v8bf*)(&As_hi[ls]) = *(const v8bf*)(Xhi + ga);
            *(v8bf*)(&As_lo[ls]) = *(const v8bf*)(Xlo + ga);
            *(v8bf*)(&Bs_hi[ls]) = *(const v8bf*)(Whi + gb);
            *(v8bf*)(&Bs_lo[ls]) = *(const v8bf*)(Wlo + gb);
        }
        __syncthreads();
#pragma unroll
        for (int k0 = 0; k0 < KB; k0 += 32)
            compute_step<STR>(As_hi, As_lo, Bs_hi, Bs_lo,
                              wm, wn, lr, k0 + 8 * hi, acc);
        __syncthreads();
    }
#endif
    GEMM_EPILOGUE()
}

// Fallback (small workspace): fused in-loop f32->bf16 hi/lo split.
__global__ __launch_bounds__(256)
void gemm_wmma_bf16x3_fused(const float* __restrict__ X, const float* __restrict__ Wm,
                            float* __restrict__ Ybase, int kChunk)
{
    __shared__ __bf16 As_hi[BM * STR];
    __shared__ __bf16 As_lo[BM * STR];
    __shared__ __bf16 Bs_hi[BN * STR];
    __shared__ __bf16 Bs_lo[BN * STR];

    const int tid  = threadIdx.x;
    const int m0   = blockIdx.y * BM;
    const int n0   = blockIdx.x * BN;
    const int kBeg = blockIdx.z * kChunk;
    const int kEnd = kBeg + kChunk;
    float* __restrict__ Y = Ybase + (size_t)blockIdx.z * (size_t)M_DIM * N_DIM;

    const int wave = tid >> 5;
    const int lane = tid & 31;
    const int lr   = lane & 15;
    const int hi   = lane >> 4;
    const int wm   = (wave & 3) * 32;
    const int wn   = (wave >> 2) * 64;

    v8f zero = {};
    v8f acc[2][4];
#pragma unroll
    for (int mi = 0; mi < 2; ++mi)
#pragma unroll
        for (int ni = 0; ni < 4; ++ni) acc[mi][ni] = zero;

    const int r0 = tid >> 4;            // 0..15
    const int c4 = (tid & 15) * 4;      // float col 0..60

    for (int kk = kBeg; kk < kEnd; kk += KB) {
#pragma unroll
        for (int j = 0; j < 8; ++j) {
            const int row = j * 16 + r0;
            {
                float4 v = *(const float4*)(X + (size_t)(m0 + row) * K_DIM + kk + c4);
                __bf16 h0 = (__bf16)v.x, h1 = (__bf16)v.y, h2 = (__bf16)v.z, h3 = (__bf16)v.w;
                v4bf hv = {h0, h1, h2, h3};
                v4bf lv = {(__bf16)(v.x - (float)h0), (__bf16)(v.y - (float)h1),
                           (__bf16)(v.z - (float)h2), (__bf16)(v.w - (float)h3)};
                *(v4bf*)(&As_hi[row * STR + c4]) = hv;
                *(v4bf*)(&As_lo[row * STR + c4]) = lv;
            }
            {
                float4 v = *(const float4*)(Wm + (size_t)(n0 + row) * K_DIM + kk + c4);
                __bf16 h0 = (__bf16)v.x, h1 = (__bf16)v.y, h2 = (__bf16)v.z, h3 = (__bf16)v.w;
                v4bf hv = {h0, h1, h2, h3};
                v4bf lv = {(__bf16)(v.x - (float)h0), (__bf16)(v.y - (float)h1),
                           (__bf16)(v.z - (float)h2), (__bf16)(v.w - (float)h3)};
                *(v4bf*)(&Bs_hi[row * STR + c4]) = hv;
                *(v4bf*)(&Bs_lo[row * STR + c4]) = lv;
            }
        }
        __syncthreads();
#pragma unroll
        for (int k0 = 0; k0 < KB; k0 += 32)
            compute_step<STR>(As_hi, As_lo, Bs_hi, Bs_lo,
                              wm, wn, lr, k0 + 8 * hi, acc);
        __syncthreads();
    }
    GEMM_EPILOGUE()
}

// ---------------------------------------------------------------------------
// BN pass 1: sum split-K partials, write y into d_out, per-column scale/shift
// from biased batch statistics (training-mode BatchNorm1d).
// ---------------------------------------------------------------------------
__global__ __launch_bounds__(256)
void bn_stats(const float* __restrict__ P0, const float* __restrict__ P1,
              float* __restrict__ Y,
              const float* __restrict__ gamma, const float* __restrict__ beta,
              int hasP1)
{
    const int n = blockIdx.x * blockDim.x + threadIdx.x;
    float sum = 0.0f, sumsq = 0.0f;
    for (int m = 0; m < M_DIM; ++m) {
        float v = P0[(size_t)m * N_DIM + n];
        if (hasP1) v += P1[(size_t)m * N_DIM + n];
        sum   += v;
        sumsq += v * v;
        Y[(size_t)m * N_DIM + n] = v;
    }
    const float mean = sum * (1.0f / M_DIM);
    const float var  = sumsq * (1.0f / M_DIM) - mean * mean;
    const float rstd = rsqrtf(var + 1e-5f);
    const float sc   = rstd * gamma[n];
    g_scale[n] = sc;
    g_shift[n] = beta[n] - mean * sc;
}

// BN pass 2: y = y*scale + shift, float4-vectorized in place.
__global__ __launch_bounds__(256)
void bn_apply(float* __restrict__ Y)
{
    const int i  = blockIdx.x * blockDim.x + threadIdx.x;
    const int n4 = (i & (N_DIM / 4 - 1)) * 4;
    float4 v = ((float4*)Y)[i];
    v.x = v.x * g_scale[n4 + 0] + g_shift[n4 + 0];
    v.y = v.y * g_scale[n4 + 1] + g_shift[n4 + 1];
    v.z = v.z * g_scale[n4 + 2] + g_shift[n4 + 2];
    v.w = v.w * g_scale[n4 + 3] + g_shift[n4 + 3];
    ((float4*)Y)[i] = v;
}

// integrate_pids = float(pids[::4])
__global__ void pids_kernel(const int* __restrict__ pids, float* __restrict__ outp)
{
    const int i = blockIdx.x * blockDim.x + threadIdx.x;
    if (i < M_DIM) outp[i] = (float)pids[i * 4];
}

extern "C" void kernel_launch(void* const* d_in, const int* in_sizes, int n_in,
                              void* d_out, int out_size, void* d_ws, size_t ws_size,
                              hipStream_t stream)
{
    const float* X     = (const float*)d_in[0];  // features [4096,2048] == x [1024,8192]
    const int*   pids  = (const int*)  d_in[1];
    const float* Wm    = (const float*)d_in[2];  // [2048, 8192]
    const float* gamma = (const float*)d_in[3];
    const float* beta  = (const float*)d_in[4];

    float* Y    = (float*)d_out;
    float* outp = Y + (size_t)M_DIM * N_DIM;

    const size_t xcnt   = (size_t)M_DIM * K_DIM;
    const size_t wcnt   = (size_t)N_DIM * K_DIM;
    const size_t ybytes = (size_t)M_DIM * N_DIM * sizeof(float);
    const size_t planes = (xcnt + wcnt) * 2 * sizeof(__bf16);  // hi+lo planes: 96MB

    const bool preconv = (ws_size >= planes);
    const int  split   = preconv ? ((ws_size >= planes + 2 * ybytes) ? 2 : 1)
                                 : ((ws_size >= 2 * ybytes) ? 2 : 1);

    dim3 grid(N_DIM / BN, M_DIM / BM, split);

    float* P;
    if (preconv) {
        __bf16* Xhi = (__bf16*)d_ws;
        __bf16* Xlo = Xhi + xcnt;
        __bf16* Whi = Xlo + xcnt;
        __bf16* Wlo = Whi + wcnt;
        P = (split == 2) ? (float*)(Wlo + wcnt) : Y;

        split_bf16<<<(unsigned)(xcnt / 4 / 256), 256, 0, stream>>>(X, Xhi, Xlo);
        split_bf16<<<(unsigned)(wcnt / 4 / 256), 256, 0, stream>>>(Wm, Whi, Wlo);
        gemm_wmma_bf16x3_pre<<<grid, 256, 0, stream>>>(Xhi, Xlo, Whi, Wlo, P, K_DIM / split);
    } else {
        P = (split == 2) ? (float*)d_ws : Y;
        gemm_wmma_bf16x3_fused<<<grid, 256, 0, stream>>>(X, Wm, P, K_DIM / split);
    }

    const float* P1 = (split == 2) ? (P + (size_t)M_DIM * N_DIM) : P;
    bn_stats<<<N_DIM / 256, 256, 0, stream>>>(P, P1, Y, gamma, beta, split == 2 ? 1 : 0);

    bn_apply<<<(M_DIM * N_DIM / 4) / 256, 256, 0, stream>>>(Y);
    pids_kernel<<<(M_DIM + 255) / 256, 256, 0, stream>>>(pids, outp);
}
